// ParallelBAttention_71073118814793
// MI455X (gfx1250) — matrix operand order repaired and checked by reading, not compiled, not run on hardware
//
#include <hip/hip_runtime.h>

typedef _Float16 half_t;
typedef __attribute__((ext_vector_type(16))) _Float16 v16h;
typedef __attribute__((ext_vector_type(8)))  float    v8f;
typedef __attribute__((ext_vector_type(4)))  unsigned int v4u;
typedef __attribute__((ext_vector_type(8)))  int v8i;
typedef __attribute__((ext_vector_type(4)))  int v4i;

#define B_  8
#define C_  256
#define V_  96
#define T_  96
#define C8_ 32
#define C2_ 128
#define P_  98          // padded spatial extent (halo of 1)
#define PP_ (P_ * P_)   // 9604

// ---------------- WMMA fragment helpers (16x16x32 f16, contiguous k order used by this file) ----------------
// A (16x32, MxK): lanes 0-15 -> M=lane, K = k0..k0+15
//                 lanes 16-31 -> M=lane-16, K = k0+16..k0+31
__device__ __forceinline__ v16h ld_afrag(const half_t* __restrict__ row, int kh) {
  v16h a;
  const half_t* p0 = row + kh * 16;
#pragma unroll
  for (int i = 0; i < 16; ++i) { a[i] = p0[i]; }
  return a;
}
// B (32x16, KxN): lanes 0-15 -> N=lane, K=k0..k0+15 ; lanes 16-31 -> N=lane-16, K=k0+16..k0+31
__device__ __forceinline__ v16h ld_bfrag(const half_t* __restrict__ p) {
  v16h b;
#pragma unroll
  for (int i = 0; i < 16; ++i) b[i] = p[i];
  return b;
}
__device__ __forceinline__ v16h ld_bfrag_strided(const half_t* p, int stride) {
  v16h b;
#pragma unroll
  for (int i = 0; i < 16; ++i) b[i] = p[i * stride];
  return b;
}
__device__ __forceinline__ v8f wmma16(v16h a, v16h b, v8f c) {
  return __builtin_amdgcn_wmma_f32_16x16x32_f16(false, a, false, b, (short)0, c, false, false);
}
// C/D (16x16 f32): lane<16: N=lane, VGPR i -> M=i ; lane>=16: N=lane-16, VGPR i -> M=8+i

// ---------------- Tensor Data Mover: async 2-D tile (global -> LDS) ----------------
// D# per CDNA5 ISA: group0 = {flags, lds_addr, global_addr, type=2},
// group1 = {data_size=2B, tensor_dim0/1, tile_dim0/1, dim0_stride}. 2-D -> groups 2/3 NULL/zero.
__device__ __forceinline__ void tdm_load_2d(const void* gaddr, unsigned lds_off,
                                            unsigned width_elems, unsigned rows,
                                            unsigned row_stride_elems) {
  unsigned long long ga = (unsigned long long)(size_t)gaddr;
  v4u g0 = { 1u,                                   // count=1 (valid user descriptor)
             lds_off,                              // LDS byte address
             (unsigned)(ga & 0xffffffffu),
             (unsigned)((ga >> 32) & 0x01ffffffu) | (2u << 30) };  // addr[56:32] | type=2
  v8i g1 = { (int)(1u << 16),                                      // data_size=1 -> 2 bytes
             (int)((width_elems & 0xffffu) << 16),                 // tensor_dim0[15:0]
             (int)(((width_elems >> 16) & 0xffffu) | ((rows & 0xffffu) << 16)), // dim0 hi | dim1 lo
             (int)(((rows >> 16) & 0xffffu) | ((width_elems & 0xffffu) << 16)), // dim1 hi | tile_dim0
             (int)(rows & 0xffffu),                                // tile_dim1 (tile_dim2=0)
             (int)row_stride_elems,                                // tensor_dim0_stride lo32
             0, 0 };
  v4i z4 = { 0, 0, 0, 0 };
#if __clang_major__ >= 23
  v8i z8 = { 0, 0, 0, 0, 0, 0, 0, 0 };
  __builtin_amdgcn_tensor_load_to_lds(g0, g1, z4, z4, z8, 0);
#else
  __builtin_amdgcn_tensor_load_to_lds(g0, g1, z4, z4, 0);
#endif
}

// ---------------- prep kernels ----------------
__global__ void k_zero4(float4* __restrict__ p) {
  p[(size_t)blockIdx.x * 256 + threadIdx.x] = float4{0.f, 0.f, 0.f, 0.f};
}

__global__ void k_prep_x(const float* __restrict__ x, float* __restrict__ out,
                         half_t* __restrict__ xh) {
  int idx = blockIdx.x * 256 + threadIdx.x;          // B*C*V*T threads exactly
  float val = x[idx];
  out[idx] = val;                                    // residual term
  int t = idx % T_; int r = idx / T_;
  int v = r % V_;  r /= V_;
  int c = r % C_;  int b = r / C_;
  xh[((size_t)(b * PP_ + (v + 1) * P_ + (t + 1))) * C_ + c] = (half_t)val;
}

__global__ void k_prepw3(const float* __restrict__ w, half_t* __restrict__ wt,
                         int O, int Cin, int total) {
  int idx = blockIdx.x * 256 + threadIdx.x;          // idx = (o*Cin + c)*9 + tap
  if (idx >= total) return;
  int tap = idx % 9; int r = idx / 9;
  int c = r % Cin; int o = r / Cin;
  wt[((size_t)tap * O + o) * Cin + c] = (half_t)w[idx];
}

__global__ void k_prepw1(const float* __restrict__ w, half_t* __restrict__ wt, int total) {
  int idx = blockIdx.x * 256 + threadIdx.x;
  if (idx < total) wt[idx] = (half_t)w[idx];
}

// ---------------- implicit-GEMM 3x3 conv (halo-padded input, no predication) ----------------
// in [B,98,98,Cin] f16 (zero halo), wt [9][Cout][Cin] f16, out [B,V,T,Cout] f16
// one wave = 16 pixels x 32 output channels (N-blocked x2)
__global__ void k_conv3_h(const half_t* __restrict__ in, const half_t* __restrict__ wt,
                          half_t* __restrict__ out, int Cin, int Cout, int nPair,
                          int totalTiles) {
  int wave = threadIdx.x >> 5, lane = threadIdx.x & 31;
  int tile = blockIdx.x * 8 + wave;
  if (tile >= totalTiles) return;
  int tileM = tile / nPair, tn = tile % nPair;
  int l15 = lane & 15, kh = lane >> 4;
  int p = tileM * 16 + l15;
  int t = p % T_; int pv = p / T_; int v = pv % V_; int b = pv / V_;
  int n0 = tn * 32 + l15, n1 = n0 + 16;
  v8f acc0 = {}, acc1 = {};
  for (int tap = 0; tap < 9; ++tap) {
    int ky = tap / 3, kx = tap % 3;
    const half_t* arow  = in + ((size_t)(b * PP_ + (v + ky) * P_ + (t + kx))) * Cin;
    const half_t* bcol0 = wt + ((size_t)(tap * Cout + n0)) * Cin;
    const half_t* bcol1 = wt + ((size_t)(tap * Cout + n1)) * Cin;
    for (int k0 = 0; k0 < Cin; k0 += 32) {
      v16h a  = ld_afrag(arow + k0, kh);
      v16h b0 = ld_bfrag(bcol0 + k0 + kh * 16);
      v16h b1 = ld_bfrag(bcol1 + k0 + kh * 16);
      acc0 = wmma16(a, b0, acc0);
      acc1 = wmma16(a, b1, acc1);
    }
  }
#pragma unroll
  for (int i = 0; i < 8; ++i) {
    int pp = tileM * 16 + kh * 8 + i;
    int t2 = pp % T_; int pv2 = pp / T_; int v2 = pv2 % V_; int b2 = pv2 / V_;
    size_t o = ((size_t)((b2 * V_ + v2) * T_ + t2)) * Cout;
    out[o + n0] = (half_t)acc0[i];
    out[o + n1] = (half_t)acc1[i];
  }
}

// ---------------- 3x3 output projection: d_out += sigma * conv(av) ----------------
// in [B,98,98,Cin] f16 (zero halo), out [B,256,V,T] f32
__global__ void k_conv3_acc(const half_t* __restrict__ in, const half_t* __restrict__ wt,
                            float* __restrict__ out, const float* __restrict__ sigma,
                            int Cin) {
  const int Cout = C_;
  int wave = threadIdx.x >> 5, lane = threadIdx.x & 31;
  int tile = blockIdx.x * 8 + wave;                  // 4608 * 8 tiles exactly
  if (tile >= 36864) return;
  int tileM = tile >> 3, tn = tile & 7;
  int l15 = lane & 15, kh = lane >> 4;
  int p = tileM * 16 + l15;
  int t = p % T_; int pv = p / T_; int v = pv % V_; int b = pv / V_;
  int n0 = tn * 32 + l15, n1 = n0 + 16;
  v8f acc0 = {}, acc1 = {};
  for (int tap = 0; tap < 9; ++tap) {
    int ky = tap / 3, kx = tap % 3;
    const half_t* arow  = in + ((size_t)(b * PP_ + (v + ky) * P_ + (t + kx))) * Cin;
    const half_t* bcol0 = wt + ((size_t)(tap * Cout + n0)) * Cin;
    const half_t* bcol1 = wt + ((size_t)(tap * Cout + n1)) * Cin;
    for (int k0 = 0; k0 < Cin; k0 += 32) {
      v16h a  = ld_afrag(arow + k0, kh);
      v16h b0 = ld_bfrag(bcol0 + k0 + kh * 16);
      v16h b1 = ld_bfrag(bcol1 + k0 + kh * 16);
      acc0 = wmma16(a, b0, acc0);
      acc1 = wmma16(a, b1, acc1);
    }
  }
  float s = sigma[0];
#pragma unroll
  for (int i = 0; i < 8; ++i) {
    int pp = tileM * 16 + kh * 8 + i;
    int t2 = pp % T_; int pv2 = pp / T_; int v2 = pv2 % V_; int b2 = pv2 / V_;
    size_t o0 = (((size_t)b2 * C_ + n0) * V_ + v2) * T_ + t2;
    size_t o1 = (((size_t)b2 * C_ + n1) * V_ + v2) * T_ + t2;
    out[o0] += s * acc0[i];                          // stream-ordered RMW, unique elements
    out[o1] += s * acc1[i];
  }
}

// ---------------- 1x1 conv for cAttn q/k/v: out [B,V,C,T] f16 (t contiguous) ----------------
__global__ void k_conv1_c(const half_t* __restrict__ xh, const half_t* __restrict__ w1,
                          const float* __restrict__ bias, half_t* __restrict__ out) {
  int wave = threadIdx.x >> 5, lane = threadIdx.x & 31;
  int tile = blockIdx.x * 8 + wave;                  // (b, v, tt, ct) = 8*96*6*16 tiles
  int ct = tile & 15; int r = tile >> 4;
  int tt = r % 6; r /= 6;
  int v = r % V_; int b = r / V_;
  int l15 = lane & 15, kh = lane >> 4;
  const half_t* arow = w1 + (size_t)(ct * 16 + l15) * C_;                     // W[c][d]
  const half_t* bcol = xh + ((size_t)(b * PP_ + (v + 1) * P_ + tt * 16 + l15 + 1)) * C_;
  v8f acc = {};
  for (int k0 = 0; k0 < C_; k0 += 32) {
    v16h a  = ld_afrag(arow + k0, kh);
    v16h bb = ld_bfrag(bcol + k0 + kh * 16);
    acc = wmma16(a, bb, acc);
  }
  size_t sb = (size_t)(b * V_ + v) * C_ * T_;
  int tcol = tt * 16 + l15;
#pragma unroll
  for (int i = 0; i < 8; ++i) {
    int c = ct * 16 + kh * 8 + i;
    out[sb + (size_t)c * T_ + tcol] = (half_t)(acc[i] + bias[c]);
  }
}

// ---------------- fused vAttn/tAttn: one workgroup per (b,t) / (b,v) slice ----------------
// q/k: [B,V,T,32] f16 ; v: [B,V,T,128] f16 ; out av: padded [B,98,98,128] f16
// dynamic LDS: sV 24576 B | sS 36864 B | sP 18432 B  (79872 B total)
__global__ void k_attn_vt(const half_t* __restrict__ qp, const half_t* __restrict__ kp,
                          const half_t* __restrict__ vp, half_t* __restrict__ op,
                          int mode) {
  extern __shared__ char smem[];
  half_t* sV = (half_t*)smem;                        // [96][128]
  float*  sS = (float*)(smem + 24576);               // [96][96]
  half_t* sP = (half_t*)(smem + 61440);              // [96][96]
  int s = blockIdx.x;
  int b = s / 96, u = s % 96;
  int qBase, qRS, vBase, vRS, oBase, oRS;
  if (mode == 0) {                   // vAttn: slice (b, t=u); rows = v
    qBase = (b * 9216 + u) * C8_;      qRS = T_ * C8_;
    vBase = (b * 9216 + u) * C2_;      vRS = T_ * C2_;
    oBase = (b * PP_ + P_ + (u + 1)) * C2_;          oRS = P_ * C2_;
  } else {                           // tAttn: slice (b, v=u); rows = t
    qBase = (b * 9216 + u * 96) * C8_; qRS = C8_;
    vBase = (b * 9216 + u * 96) * C2_; vRS = C2_;
    oBase = (b * PP_ + (u + 1) * P_ + 1) * C2_;      oRS = C2_;
  }
  int wave = threadIdx.x >> 5, lane = threadIdx.x & 31;
  int l15 = lane & 15, kh = lane >> 4;

  // kick off async TDM: V slice (96 rows x 128 elems, row stride vRS) -> LDS @0
  if (wave == 0) tdm_load_2d(vp + vBase, 0u, (unsigned)C2_, 96u, (unsigned)vRS);

  // S = Q K^T  (96x96, K=32 -> single WMMA per tile), overlapped with the TDM
  for (int tile = wave; tile < 36; tile += 8) {
    int ti = tile / 6, tj = tile % 6;
    v16h a  = ld_afrag(qp + qBase + (ti * 16 + l15) * qRS, kh);
    v16h bb = ld_bfrag(kp + qBase + (tj * 16 + l15) * qRS + kh * 16);
    v8f acc = {};
    acc = wmma16(a, bb, acc);
#pragma unroll
    for (int i = 0; i < 8; ++i)
      sS[(ti * 16 + kh * 8 + i) * 96 + tj * 16 + l15] = acc[i];
  }
  __syncthreads();

  // row softmax
  if (threadIdx.x < 96) {
    int r = threadIdx.x;
    float mx = -3.0e38f;
    for (int w = 0; w < 96; ++w) mx = fmaxf(mx, sS[r * 96 + w]);
    float sum = 0.f;
    for (int w = 0; w < 96; ++w) { float e = __expf(sS[r * 96 + w] - mx); sS[r * 96 + w] = e; sum += e; }
    float inv = 1.f / sum;
    for (int w = 0; w < 96; ++w) sP[r * 96 + w] = (half_t)(sS[r * 96 + w] * inv);
  }
  if (wave == 0) __builtin_amdgcn_s_wait_tensorcnt((short)0);   // V tile landed in LDS
  __syncthreads();

  // AV = P V  (96x128, K=96), V read from LDS
  for (int tile = wave; tile < 48; tile += 8) {
    int ti = tile / 8, cj = tile % 8;
    v8f acc = {};
    for (int ks = 0; ks < 96; ks += 32) {
      v16h a  = ld_afrag(sP + (ti * 16 + l15) * 96 + ks, kh);
      v16h bb = ld_bfrag_strided(sV + (ks + kh * 16) * C2_ + cj * 16 + l15, C2_);
      acc = wmma16(a, bb, acc);
    }
#pragma unroll
    for (int i = 0; i < 8; ++i)
      op[oBase + (ti * 16 + kh * 8 + i) * oRS + cj * 16 + l15] = (half_t)acc[i];
  }
}

// ---------------- fused cAttn: one workgroup per (b,v); V staged once via TDM ----------------
// q/k/v: [B,V,C,T] f16 ; out av: padded [B,98,98,256] f16
// dynamic LDS: sV 49152 B | sS 16384 B | sP 8192 B  (73728 B total)
__global__ void k_attn_c(const half_t* __restrict__ qp, const half_t* __restrict__ kp,
                         const half_t* __restrict__ vp, half_t* __restrict__ op) {
  extern __shared__ char smem[];
  half_t* sV = (half_t*)smem;                        // [256][96]
  float*  sS = (float*)(smem + 49152);               // [16][256]
  half_t* sP = (half_t*)(smem + 65536);              // [16][256]
  int idx = blockIdx.x;
  int v = idx % V_; int b = idx / V_;
  size_t sb = (size_t)(b * V_ + v) * C_ * T_;
  int wave = threadIdx.x >> 5, lane = threadIdx.x & 31;
  int l15 = lane & 15, kh = lane >> 4;

  // stage whole V slice (contiguous 256*96 f16) into LDS asynchronously
  if (wave == 0) tdm_load_2d(vp + sb, 0u, (unsigned)(C_ * T_), 1u, (unsigned)(C_ * T_));

  for (int cb = 0; cb < 16; ++cb) {
    // S[16,256] = Qblk Kall^T over K=t (96)
    for (int nt = wave; nt < 16; nt += 8) {
      const half_t* arow = qp + sb + (size_t)(cb * 16 + l15) * T_;
      const half_t* bcol = kp + sb + (size_t)(nt * 16 + l15) * T_;
      v8f acc = {};
      for (int ks = 0; ks < 96; ks += 32) {
        v16h a  = ld_afrag(arow + ks, kh);
        v16h bb = ld_bfrag(bcol + ks + kh * 16);
        acc = wmma16(a, bb, acc);
      }
#pragma unroll
      for (int i = 0; i < 8; ++i)
        sS[(kh * 8 + i) * 256 + nt * 16 + l15] = acc[i];
    }
    __syncthreads();

    if (threadIdx.x < 16) {
      int rr = threadIdx.x;
      float mx = -3.0e38f;
      for (int w = 0; w < 256; ++w) mx = fmaxf(mx, sS[rr * 256 + w]);
      float sum = 0.f;
      for (int w = 0; w < 256; ++w) { float e = __expf(sS[rr * 256 + w] - mx); sS[rr * 256 + w] = e; sum += e; }
      float inv = 1.f / sum;
      for (int w = 0; w < 256; ++w) sP[rr * 256 + w] = (half_t)(sS[rr * 256 + w] * inv);
    }
    if (cb == 0 && wave == 0) __builtin_amdgcn_s_wait_tensorcnt((short)0);
    __syncthreads();

    // AV[16,96] = P V over K=d (256), V from LDS
    for (int nt = wave; nt < 6; nt += 8) {
      int tcol = nt * 16 + l15;
      v8f acc = {};
      for (int kd = 0; kd < 256; kd += 32) {
        v16h a  = ld_afrag(sP + l15 * 256 + kd, kh);
        v16h bb = ld_bfrag_strided(sV + (kd + kh * 16) * T_ + tcol, T_);
        acc = wmma16(a, bb, acc);
      }
#pragma unroll
      for (int i = 0; i < 8; ++i) {
        int c = cb * 16 + kh * 8 + i;
        op[((size_t)(b * PP_ + (v + 1) * P_ + tcol + 1)) * C_ + c] = (half_t)acc[i];
      }
    }
    __syncthreads();   // sS/sP reused next iteration
  }
}

// ---------------- host launcher ----------------
extern "C" void kernel_launch(void* const* d_in, const int* in_sizes, int n_in,
                              void* d_out, int out_size, void* d_ws, size_t ws_size,
                              hipStream_t stream) {
  (void)in_sizes; (void)n_in; (void)out_size; (void)ws_size;
  const float* x     = (const float*)d_in[0];
  const float* v_wq  = (const float*)d_in[1];
  const float* v_wk  = (const float*)d_in[2];
  const float* v_wv  = (const float*)d_in[3];
  const float* v_wo  = (const float*)d_in[4];
  const float* v_sig = (const float*)d_in[5];
  const float* t_wq  = (const float*)d_in[6];
  const float* t_wk  = (const float*)d_in[7];
  const float* t_wv  = (const float*)d_in[8];
  const float* t_wo  = (const float*)d_in[9];
  const float* t_sig = (const float*)d_in[10];
  const float* c_wq  = (const float*)d_in[11];
  const float* c_bq  = (const float*)d_in[12];
  const float* c_wk  = (const float*)d_in[13];
  const float* c_bk  = (const float*)d_in[14];
  const float* c_wv  = (const float*)d_in[15];
  const float* c_bv  = (const float*)d_in[16];
  const float* c_wo  = (const float*)d_in[17];
  const float* c_sig = (const float*)d_in[18];
  float* out = (float*)d_out;

  char* ws = (char*)d_ws;
  auto carve = [&](size_t halves) -> half_t* {
    half_t* p = (half_t*)ws;
    ws += ((halves * sizeof(half_t) + 255) & ~(size_t)255);
    return p;
  };
  const size_t NPIX = (size_t)B_ * V_ * T_;          // 73728
  const size_t NPAD = (size_t)B_ * PP_;              // 76832 padded pixels
  half_t* XH    = carve(NPAD * C_);                  // x channels-last f16, zero halo
  half_t* WT_VQ = carve(9 * 32 * 256);
  half_t* WT_VK = carve(9 * 32 * 256);
  half_t* WT_VV = carve(9 * 128 * 256);
  half_t* WT_VO = carve(9 * 256 * 128);
  half_t* WT_TQ = carve(9 * 32 * 256);
  half_t* WT_TK = carve(9 * 32 * 256);
  half_t* WT_TV = carve(9 * 128 * 256);
  half_t* WT_TO = carve(9 * 256 * 128);
  half_t* WT_CO = carve(9 * 256 * 256);
  half_t* W1Q   = carve(256 * 256);
  half_t* W1K   = carve(256 * 256);
  half_t* W1V   = carve(256 * 256);
  half_t* R0 = carve(NPIX * C_);                     // Q
  half_t* R1 = carve(NPIX * C_);                     // K
  half_t* R2 = carve(NPIX * C_);                     // V
  half_t* R3 = carve(NPAD * C_);                     // AV (padded, zero halo)

  // zero the padded buffers, residual copy + f16 transpose
  k_zero4<<<9604, 256, 0, stream>>>((float4*)XH);    // 8*9604*256 halves = 9604*256 float4
  k_prep_x<<<73728, 256, 0, stream>>>(x, out, XH);

  auto w3 = [&](const float* w, half_t* wt, int O, int Cin) {
    int tot = O * Cin * 9;
    k_prepw3<<<(tot + 255) / 256, 256, 0, stream>>>(w, wt, O, Cin, tot);
  };
  w3(v_wq, WT_VQ, 32, 256);  w3(v_wk, WT_VK, 32, 256);
  w3(v_wv, WT_VV, 128, 256); w3(v_wo, WT_VO, 256, 128);
  w3(t_wq, WT_TQ, 32, 256);  w3(t_wk, WT_TK, 32, 256);
  w3(t_wv, WT_TV, 128, 256); w3(t_wo, WT_TO, 256, 128);
  w3(c_wo, WT_CO, 256, 256);
  k_prepw1<<<256, 256, 0, stream>>>(c_wq, W1Q, 256 * 256);
  k_prepw1<<<256, 256, 0, stream>>>(c_wk, W1K, 256 * 256);
  k_prepw1<<<256, 256, 0, stream>>>(c_wv, W1V, 256 * 256);

  // -------- vAttn branch --------
  k_zero4<<<9604, 256, 0, stream>>>((float4*)R3);    // zero halo for 128-ch layout
  k_conv3_h<<<576,  256, 0, stream>>>(XH, WT_VQ, R0, 256, 32, 1, 4608);
  k_conv3_h<<<576,  256, 0, stream>>>(XH, WT_VK, R1, 256, 32, 1, 4608);
  k_conv3_h<<<2304, 256, 0, stream>>>(XH, WT_VV, R2, 256, 128, 4, 18432);
  k_attn_vt<<<768, 256, 79872, stream>>>(R0, R1, R2, R3, 0);
  k_conv3_acc<<<4608, 256, 0, stream>>>(R3, WT_VO, out, v_sig, 128);

  // -------- tAttn branch (same padded layout -> halo still zero) --------
  k_conv3_h<<<576,  256, 0, stream>>>(XH, WT_TQ, R0, 256, 32, 1, 4608);
  k_conv3_h<<<576,  256, 0, stream>>>(XH, WT_TK, R1, 256, 32, 1, 4608);
  k_conv3_h<<<2304, 256, 0, stream>>>(XH, WT_TV, R2, 256, 128, 4, 18432);
  k_attn_vt<<<768, 256, 79872, stream>>>(R0, R1, R2, R3, 1);
  k_conv3_acc<<<4608, 256, 0, stream>>>(R3, WT_TO, out, t_sig, 128);

  // -------- cAttn branch (256-ch layout: re-zero halo) --------
  k_conv1_c<<<9216, 256, 0, stream>>>(XH, W1Q, c_bq, R0);
  k_conv1_c<<<9216, 256, 0, stream>>>(XH, W1K, c_bk, R1);
  k_conv1_c<<<9216, 256, 0, stream>>>(XH, W1V, c_bv, R2);
  k_zero4<<<9604, 256, 0, stream>>>((float4*)R3);
  k_attn_c<<<768, 256, 73728, stream>>>(R0, R1, R2, R3);
  k_conv3_acc<<<4608, 256, 0, stream>>>(R3, WT_CO, out, c_sig, 256);
}
